// GCN_74500502716954
// MI455X (gfx1250) — matrix operand order repaired and checked by reading, not compiled, not run on hardware
//
#include <hip/hip_runtime.h>
#include <hip/hip_bf16.h>

typedef __bf16 v16bf __attribute__((ext_vector_type(16)));
typedef float  v8f   __attribute__((ext_vector_type(8)));

union FragBF {
    v16bf v;
    __bf16 e[16];
    unsigned short u[16];
    uint4 q[2];
};

// Native conversion: lowers to v_cvt_pk_bf16_f32 / v_cvt_bf16_f32 on gfx1250
// (round-to-nearest-even), instead of a 3-VALU-op bit-twiddle per element.
__device__ __forceinline__ __bf16 to_bf16(float f) { return (__bf16)f; }

__device__ __forceinline__ unsigned short f32_to_bf16_bits(float f) {
    unsigned int x = __float_as_uint(f);
    unsigned int r = x + 0x7FFFu + ((x >> 16) & 1u);   // RNE
    return (unsigned short)(r >> 16);
}

__device__ __forceinline__ bool keep_mask(int m, int k) {
    unsigned h = ((unsigned)m * 73856093u) ^ ((unsigned)k * 19349663u);
    h *= 2654435761u;
    return (h >> 16) & 1u;
}

// ---------------------------------------------------------------------------
// Pack W (row-major K x Ncols, fp32) into WMMA B-fragment order, bf16.
// Fragment order: idx = ((s*ntiles + t)*32 + lane)*16 + j
//   n = t*16 + (lane & 15);   k = s*32 + (lane>>4)*8 + (j&7) + (j>>3)*16
// Per-lane 16 halfs = 32B contiguous -> gemm loads 2x b128, fully coalesced.
// ---------------------------------------------------------------------------
__global__ void pack_w_frag(const float* __restrict__ W, unsigned short* __restrict__ frag,
                            int K, int Ncols, int ksteps, int ntiles) {
    int idx = blockIdx.x * blockDim.x + threadIdx.x;
    int total = ksteps * ntiles * 32 * 16;
    if (idx >= total) return;
    int j    = idx & 15;
    int lane = (idx >> 4) & 31;
    int rest = idx >> 9;            // s*ntiles + t
    int t = rest % ntiles;
    int s = rest / ntiles;
    int n = t * 16 + (lane & 15);
    int k = s * 32 + (lane >> 4) * 8 + (j & 7) + ((j >> 3) << 4);
    float v = (k < K && n < Ncols) ? W[(size_t)k * Ncols + n] : 0.0f;
    frag[idx] = f32_to_bf16_bits(v);
}

// ---------------------------------------------------------------------------
// GEMM1: H0[N x 64] = x[N x 512] @ W1, bf16 WMMA, fp32 accumulate.
// One wave per 16-row tile; 4 N-tiles of 16; 16 K-steps of 32.
// A-fragment (16x32 bf16, per lane): two contiguous 8-float runs:
//   k = s*32 + hi*8 + j            (j = 0..7)
//   k = s*32 + 16 + hi*8 + (j-8)   (j = 8..15)
// ---------------------------------------------------------------------------
__global__ void gemm1_wmma(const float* __restrict__ x, const unsigned short* __restrict__ w1f,
                           float* __restrict__ H0, int Nnodes) {
    const int lane  = threadIdx.x & 31;
    const int wave  = threadIdx.x >> 5;
    const int mtile = blockIdx.x * 4 + wave;
    if (mtile * 16 >= Nnodes) return;
    const int hi   = lane >> 4;
    const int lmod = lane & 15;
    int m = mtile * 16 + lmod;
    if (m >= Nnodes) m = Nnodes - 1;

    const float* xrow = x + (size_t)m * 512 + hi * 8;
    const uint4* bbase = (const uint4*)w1f;

    v8f acc[4] = {};
#pragma unroll
    for (int s = 0; s < 16; ++s) {
        const float4* p = (const float4*)(xrow + s * 32);
        float4 f0 = p[0], f1 = p[1], f2 = p[4], f3 = p[5];
        float a16[16] = { f0.x, f0.y, f0.z, f0.w, f1.x, f1.y, f1.z, f1.w,
                          f2.x, f2.y, f2.z, f2.w, f3.x, f3.y, f3.z, f3.w };
        FragBF a;
#pragma unroll
        for (int j = 0; j < 16; ++j) a.e[j] = to_bf16(a16[j]);

#pragma unroll
        for (int t = 0; t < 4; ++t) {
            FragBF b;
            const uint4* bp = bbase + (((size_t)(s * 4 + t) * 32 + lane) * 2);
            b.q[0] = bp[0];
            b.q[1] = bp[1];
            acc[t] = __builtin_amdgcn_wmma_f32_16x16x32_bf16(
                false, a.v, false, b.v, (short)0, acc[t], false, false);
        }
    }

    // D layout: VGPR r, lane -> M = r + hi*8, N = lane & 15
#pragma unroll
    for (int t = 0; t < 4; ++t) {
#pragma unroll
        for (int r = 0; r < 8; ++r) {
            int mm = mtile * 16 + hi * 8 + r;
            if (mm < Nnodes) H0[(size_t)mm * 64 + t * 16 + lmod] = acc[t][r];
        }
    }
}

// ---------------------------------------------------------------------------
// GEMM2: G[N x 40] = post(S1) @ W2, where post = relu(S1 + b1) * dropmask * 2.
// bias/relu/dropout fused into A-fragment conversion. K = 64 (2 steps), 3 N-tiles.
// ---------------------------------------------------------------------------
__global__ void gemm2_wmma(const float* __restrict__ S1, const float* __restrict__ b1,
                           const unsigned short* __restrict__ w2f,
                           float* __restrict__ G, int Nnodes) {
    const int lane  = threadIdx.x & 31;
    const int wave  = threadIdx.x >> 5;
    const int mtile = blockIdx.x * 4 + wave;
    if (mtile * 16 >= Nnodes) return;
    const int hi   = lane >> 4;
    const int lmod = lane & 15;
    int m = mtile * 16 + lmod;
    if (m >= Nnodes) m = Nnodes - 1;

    const float* hrow = S1 + (size_t)m * 64 + hi * 8;
    const uint4* bbase = (const uint4*)w2f;

    v8f acc[3] = {};
#pragma unroll
    for (int s = 0; s < 2; ++s) {
        const float4* p = (const float4*)(hrow + s * 32);
        float4 f0 = p[0], f1 = p[1], f2 = p[4], f3 = p[5];
        float a16[16] = { f0.x, f0.y, f0.z, f0.w, f1.x, f1.y, f1.z, f1.w,
                          f2.x, f2.y, f2.z, f2.w, f3.x, f3.y, f3.z, f3.w };
        const int k0 = s * 32 + hi * 8;
        FragBF a;
#pragma unroll
        for (int j = 0; j < 16; ++j) {
            int k = k0 + j + ((j >= 8) ? 8 : 0);
            float v = a16[j] + b1[k];
            v = v > 0.0f ? v : 0.0f;
            v = keep_mask(m, k) ? v * 2.0f : 0.0f;   // dropout p=0.5, 1/(1-p)=2
            a.e[j] = to_bf16(v);
        }
#pragma unroll
        for (int t = 0; t < 3; ++t) {
            FragBF b;
            const uint4* bp = bbase + (((size_t)(s * 3 + t) * 32 + lane) * 2);
            b.q[0] = bp[0];
            b.q[1] = bp[1];
            acc[t] = __builtin_amdgcn_wmma_f32_16x16x32_bf16(
                false, a.v, false, b.v, (short)0, acc[t], false, false);
        }
    }

#pragma unroll
    for (int t = 0; t < 3; ++t) {
        int n = t * 16 + lmod;
        if (n >= 40) continue;
#pragma unroll
        for (int r = 0; r < 8; ++r) {
            int mm = mtile * 16 + hi * 8 + r;
            if (mm < Nnodes) G[(size_t)mm * 40 + n] = acc[t][r];
        }
    }
}

// ---------------------------------------------------------------------------
// COO SpMM scatter: out[row[e]] += vals[e] * dense[col[e]], F features.
// One thread per (edge, 4-feature chunk); float4 gather, 4x f32 atomic scatter.
// Gather table (<= 25.6 MB) is L2-resident on the 192 MB L2.
// ---------------------------------------------------------------------------
__global__ void spmm_scatter(const int* __restrict__ row, const int* __restrict__ col,
                             const float* __restrict__ vals, const float* __restrict__ dense,
                             float* __restrict__ out, int E, int F) {
    int idx = blockIdx.x * blockDim.x + threadIdx.x;
    int e  = idx >> 4;
    int ch = idx & 15;
    int nch = F >> 2;
    if (e >= E || ch >= nch) return;
    int c = col[e];
    int r = row[e];
    float v = vals[e];
    float4 d4 = *(const float4*)(dense + (size_t)c * F + ch * 4);
    float* o = out + (size_t)r * F + ch * 4;
    atomicAdd(o + 0, v * d4.x);
    atomicAdd(o + 1, v * d4.y);
    atomicAdd(o + 2, v * d4.z);
    atomicAdd(o + 3, v * d4.w);
}

// ---------------------------------------------------------------------------
// Per-row (40 classes) bias + log_softmax, one wave32 per row.
// lane handles element lane (always valid, 40>32) and lane+32 (lanes 0..7).
// ---------------------------------------------------------------------------
__global__ void logsoftmax40(float* __restrict__ out, const float* __restrict__ b2, int Nnodes) {
    int gwave = (blockIdx.x * blockDim.x + threadIdx.x) >> 5;
    int lane  = threadIdx.x & 31;
    if (gwave >= Nnodes) return;
    float* rowp = out + (size_t)gwave * 40;

    float v0 = rowp[lane] + b2[lane];
    float v1 = (lane < 8) ? (rowp[32 + lane] + b2[32 + lane]) : -3.402823466e38f;

    float mx = fmaxf(v0, v1);
#pragma unroll
    for (int o = 16; o > 0; o >>= 1) mx = fmaxf(mx, __shfl_xor(mx, o, 32));

    float s = __expf(v0 - mx) + ((lane < 8) ? __expf(v1 - mx) : 0.0f);
#pragma unroll
    for (int o = 16; o > 0; o >>= 1) s += __shfl_xor(s, o, 32);

    float lse = __logf(s) + mx;
    rowp[lane] = v0 - lse;
    if (lane < 8) rowp[32 + lane] = v1 - lse;
}

// ---------------------------------------------------------------------------
// Inputs (setup_inputs order):
//  0: x [N,512] f32   1: edge_vals [E] f32   2: W1 [512,64] f32  3: b1 [64]
//  4: W2 [64,40] f32  5: b2 [40] f32         6: edge_row [E] i32 7: edge_col [E] i32
// Output: [N,40] f32 log_softmax.
// ---------------------------------------------------------------------------
extern "C" void kernel_launch(void* const* d_in, const int* in_sizes, int n_in,
                              void* d_out, int out_size, void* d_ws, size_t ws_size,
                              hipStream_t stream) {
    const float* x    = (const float*)d_in[0];
    const float* ev   = (const float*)d_in[1];
    const float* W1   = (const float*)d_in[2];
    const float* b1   = (const float*)d_in[3];
    const float* W2   = (const float*)d_in[4];
    const float* b2   = (const float*)d_in[5];
    const int*   erow = (const int*)d_in[6];
    const int*   ecol = (const int*)d_in[7];
    float* out = (float*)d_out;

    const int N = in_sizes[0] / 512;
    const int E = in_sizes[1];

    // Workspace layout
    float* H0 = (float*)d_ws;                          // N*64  (x @ W1)
    float* S1 = H0 + (size_t)N * 64;                   // N*64  (spmm1 accumulator)
    float* G  = S1 + (size_t)N * 64;                   // N*40  (post(h) @ W2)
    unsigned short* w1f = (unsigned short*)(G + (size_t)N * 40);  // 16*4*32*16 halfs
    unsigned short* w2f = w1f + 16 * 4 * 32 * 16;                 //  2*3*32*16 halfs

    // Zero scatter accumulators (d_ws/d_out are poisoned, not re-zeroed).
    hipMemsetAsync(S1, 0, (size_t)N * 64 * sizeof(float), stream);
    hipMemsetAsync(out, 0, (size_t)N * 40 * sizeof(float), stream);

    // Pack weights into WMMA B-fragment order (bf16).
    {
        int tot1 = 16 * 4 * 32 * 16;
        int tot2 = 2 * 3 * 32 * 16;
        pack_w_frag<<<(tot1 + 255) / 256, 256, 0, stream>>>(W1, w1f, 512, 64, 16, 4);
        pack_w_frag<<<(tot2 + 255) / 256, 256, 0, stream>>>(W2, w2f, 64, 40, 2, 3);
    }

    const int mtiles  = (N + 15) / 16;
    const int gblocks = (mtiles + 3) / 4;      // 4 waves (16-row tiles) per block

    // H0 = x @ W1
    gemm1_wmma<<<gblocks, 128, 0, stream>>>(x, w1f, H0, N);

    // S1 = A @ H0  (COO scatter)
    {
        long long work = (long long)E * 16;
        spmm_scatter<<<(int)((work + 255) / 256), 256, 0, stream>>>(erow, ecol, ev, H0, S1, E, 64);
    }

    // G = relu(S1 + b1) * drop / 0.5 @ W2   (fused into A-fragment load)
    gemm2_wmma<<<gblocks, 128, 0, stream>>>(S1, b1, w2f, G, N);

    // out = A @ G  (COO scatter)
    {
        long long work = (long long)E * 16;
        spmm_scatter<<<(int)((work + 255) / 256), 256, 0, stream>>>(erow, ecol, ev, G, out, E, 40);
    }

    // out = log_softmax(out + b2)
    {
        long long threads = (long long)N * 32;
        logsoftmax40<<<(int)((threads + 255) / 256), 256, 0, stream>>>(out, b2, N);
    }
}
